// QLinear_3315714752753
// MI455X (gfx1250) — matrix operand order repaired and checked by reading, not compiled, not run on hardware
//
#include <hip/hip_runtime.h>
#include <hip/hip_bf16.h>

typedef __bf16 bf16_t;
typedef __attribute__((ext_vector_type(8)))  __bf16 v8bf;
typedef __attribute__((ext_vector_type(16))) __bf16 v16bf;
typedef __attribute__((ext_vector_type(8)))  float  v8f;
typedef __attribute__((ext_vector_type(4)))  int    v4i;

#define M_TOTAL 8192          // B * S
#define N_TOTAL 11008         // OUT
#define K_TOTAL 4096          // IN
#define QCOLS   (K_TOTAL / 8) // packed int32 columns per output row
#define LDS_STRIDE 40         // 32 bf16 + 8 pad -> 80B rows (16B aligned)

#if __has_builtin(__builtin_amdgcn_global_load_async_to_lds_b128) && \
    __has_builtin(__builtin_amdgcn_s_wait_asynccnt)
#define USE_ASYNC_LDS 1
typedef __attribute__((address_space(1))) v4i* gas_p;   // global b128 chunk
typedef __attribute__((address_space(3))) v4i* las_p;   // LDS b128 chunk
#else
#define USE_ASYNC_LDS 0
#endif

// 256 threads = 8 wave32. Block tile 128(M) x 128(N).
// Wave grid 2(M) x 4(N): each wave computes 64x32 = 4x2 tiles of 16x16.
// A tile (128x32 bf16) staged in LDS, double-buffered, shared by all waves.
__global__ __launch_bounds__(256) void qlinear_wmma_bf16(
    const bf16_t* __restrict__ A,   // [M, K] bf16 activations
    const int*    __restrict__ Q,   // [N, K/8] packed int4 weights
    const float*  __restrict__ Sc,  // [N] per-row scale
    bf16_t*       __restrict__ Out) // [M, N] bf16
{
  __shared__ __align__(16) bf16_t aTile[2][128 * LDS_STRIDE]; // 2 x 10KB

  const int lane   = threadIdx.x & 31;
  const int wave   = threadIdx.x >> 5;
  const int lane16 = lane & 15;
  const int laneHi = lane >> 4;

  const int mBlock = blockIdx.x * 128;
  const int nBlock = blockIdx.y * 128;
  const int mWave  = mBlock + (wave >> 2) * 64;
  const int nWave  = nBlock + (wave & 3) * 32;

  // Cooperative A-tile copy: 128 rows x 64B = 512 b128 chunks, 2 per thread.
  auto loadA = [&](int buf, int kb) {
#pragma unroll
    for (int cc = 0; cc < 2; ++cc) {
      const int c   = threadIdx.x + cc * 256;
      const int row = c >> 2;
      const int seg = (c & 3) * 8;                 // bf16 elements
      const bf16_t* g = A + (size_t)(mBlock + row) * K_TOTAL + kb + seg;
      bf16_t*       l = &aTile[buf][row * LDS_STRIDE + seg];
#if USE_ASYNC_LDS
      __builtin_amdgcn_global_load_async_to_lds_b128((gas_p)g, (las_p)l, 0, 0);
#else
      *(uint4*)l = *(const uint4*)g;               // global b128 -> ds_store_b128
#endif
    }
  };

  // B-side per-lane state (B 32x16: lane n holds col n, K {0..7,16..23} / {8..15,24..31} by lane half).
  int        oCol[2];
  float      sCol[2];
  const int* qRow[2];
#pragma unroll
  for (int j = 0; j < 2; ++j) {
    oCol[j] = nWave + j * 16 + lane16;
    sCol[j] = Sc[oCol[j]];
    qRow[j] = Q + (size_t)oCol[j] * QCOLS;
  }

  // A fragment LDS row offsets (row within 128-row block tile).
  int aOff[4];
#pragma unroll
  for (int i = 0; i < 4; ++i)
    aOff[i] = ((wave >> 2) * 64 + i * 16 + lane16) * LDS_STRIDE;

  loadA(0, 0);
#if USE_ASYNC_LDS
  __builtin_amdgcn_s_wait_asynccnt(0);
#endif
  __syncthreads();

  v8f acc[4][2] = {};

  for (int kb = 0; kb < K_TOTAL; kb += 32) {
    const int cur = (kb >> 5) & 1;
    if (kb + 32 < K_TOTAL) loadA(cur ^ 1, kb + 32);     // prefetch next tile

    __builtin_prefetch((const void*)(qRow[0] + (kb >> 3) + 16), 0, 0);

    // A frags from LDS: lane L<16 = row L, K {0..7,16..23}; lane L+16 same
    // row, K {8..15,24..31} -> two ds_load_b128 per fragment.
    union { v16bf v; v8bf h[2]; } a[4];
#pragma unroll
    for (int i = 0; i < 4; ++i) {
      const bf16_t* l = &aTile[cur][aOff[i] + laneHi * 8];
      a[i].h[0] = *(const v8bf*)l;
      a[i].h[1] = *(const v8bf*)(l + 16);
    }

    // B frags: two b32 loads of packed int4 per tile, dequant in registers.
    const int kq = (kb >> 3) + laneHi;
    v16bf b[2];
#pragma unroll
    for (int j = 0; j < 2; ++j) {
      const int   q0 = qRow[j][kq];
      const int   q1 = qRow[j][kq + 2];
      const float s  = sCol[j];
#pragma unroll
      for (int t = 0; t < 8; ++t) {
        const int n0 = (q0 << (28 - 4 * t)) >> 28;       // sign-extended nibble
        const int n1 = (q1 << (28 - 4 * t)) >> 28;
        b[j][t]     = (bf16_t)((float)n0 * s);
        b[j][t + 8] = (bf16_t)((float)n1 * s);
      }
    }

#pragma unroll
    for (int i = 0; i < 4; ++i)
#pragma unroll
      for (int j = 0; j < 2; ++j)
        acc[i][j] = __builtin_amdgcn_wmma_f32_16x16x32_bf16(
            false, a[i].v, false, b[j], (short)0, acc[i][j], false, false);

#if USE_ASYNC_LDS
    __builtin_amdgcn_s_wait_asynccnt(0);
#endif
    __syncthreads();
  }

  // ---- Store: transpose each wave's 16x32 f32 sub-tile through LDS so each
  // lane emits 32B of consecutive bf16 (2 x global_store_b128).
  __syncthreads();                                       // A-tile LDS now free
  float* cbuf = (float*)(&aTile[0][0]) + wave * 512;     // 16x32 f32 per wave
  const int crRow = lane >> 1;
  const int crCol = (lane & 1) * 16;

#pragma unroll
  for (int i = 0; i < 4; ++i) {
    // D layout: VGPR r -> row laneHi*8+r, col lane16 (per 16x16 tile j).
#pragma unroll
    for (int j = 0; j < 2; ++j)
#pragma unroll
      for (int r = 0; r < 8; ++r)
        cbuf[(laneHi * 8 + r) * 32 + j * 16 + lane16] = acc[i][j][r];

    // Wave-private region; LDS ops are in-order within a wave.
    const float4* cp = (const float4*)&cbuf[crRow * 32 + crCol];
    const float4 f0 = cp[0], f1 = cp[1], f2 = cp[2], f3 = cp[3];

    union { bf16_t h[16]; uint4 q[2]; } o;
    o.h[0]=(bf16_t)f0.x; o.h[1]=(bf16_t)f0.y; o.h[2]=(bf16_t)f0.z; o.h[3]=(bf16_t)f0.w;
    o.h[4]=(bf16_t)f1.x; o.h[5]=(bf16_t)f1.y; o.h[6]=(bf16_t)f1.z; o.h[7]=(bf16_t)f1.w;
    o.h[8]=(bf16_t)f2.x; o.h[9]=(bf16_t)f2.y; o.h[10]=(bf16_t)f2.z; o.h[11]=(bf16_t)f2.w;
    o.h[12]=(bf16_t)f3.x; o.h[13]=(bf16_t)f3.y; o.h[14]=(bf16_t)f3.z; o.h[15]=(bf16_t)f3.w;

    bf16_t* outp = Out + (size_t)(mWave + i * 16 + crRow) * N_TOTAL + nWave + crCol;
    *(uint4*)(outp)     = o.q[0];
    *(uint4*)(outp + 8) = o.q[1];
  }
}

extern "C" void kernel_launch(void* const* d_in, const int* in_sizes, int n_in,
                              void* d_out, int out_size, void* d_ws, size_t ws_size,
                              hipStream_t stream) {
  const bf16_t* A  = (const bf16_t*)d_in[0];  // input  [4,2048,4096] bf16
  const int*    Q  = (const int*)d_in[1];     // qweight [11008,512] int32
  const float*  Sc = (const float*)d_in[2];   // scale  [11008] f32
  bf16_t*       O  = (bf16_t*)d_out;          // out    [4,2048,11008] bf16

  dim3 grid(M_TOTAL / 128, N_TOTAL / 128);    // 64 x 86 blocks
  qlinear_wmma_bf16<<<grid, dim3(256), 0, stream>>>(A, Q, Sc, O);
}
